// DinoSDRTokenizer_83159156785674
// MI455X (gfx1250) — matrix operand order repaired and checked by reading, not compiled, not run on hardware
//
#include <hip/hip_runtime.h>
#include <hip/hip_bf16.h>

typedef __bf16 bf16_t;
typedef bf16_t v8bf  __attribute__((ext_vector_type(8)));
typedef bf16_t v16bf __attribute__((ext_vector_type(16)));
typedef float  v8f   __attribute__((ext_vector_type(8)));
typedef unsigned short u16;

#define N_TOK   7200      // B*L
#define CDIM    768
#define MIDD    256
#define KCB     8912
#define ROW_TILES 450     // N_TOK/16
#define COL_TILES 557     // KCB/16
#define TT2     (2.0f/0.07f)   // 2/TEMP
#define TE2     200.0f         // 2/ENT_TEMP  (ENT = TEMP/7 -> e_ent = e_temp^7)

__device__ __forceinline__ u16 f2bf(float f) {
  unsigned u = __builtin_bit_cast(unsigned, f);
  u += 0x7FFFu + ((u >> 16) & 1u);           // round-to-nearest-even
  return (u16)(u >> 16);
}
__device__ __forceinline__ v16bf cat16(v8bf lo, v8bf hi) {
  return __builtin_shufflevector(lo, hi, 0,1,2,3,4,5,6,7,8,9,10,11,12,13,14,15);
}
__device__ __forceinline__ v8f wmma_bf16(v16bf a, v16bf b, v8f c) {
  return __builtin_amdgcn_wmma_f32_16x16x32_bf16(false, a, false, b, (short)0, c, false, false);
}
__device__ __forceinline__ float pow7(float x) { float x2 = x*x; float x4 = x2*x2; return x4*x2*x; }

// ---------------- init: zero accumulators ----------------
__global__ __launch_bounds__(256) void k_init(float* avg_acc, int* occ, float* scal) {
  int id = blockIdx.x * 256 + threadIdx.x;
  if (id < KCB) { avg_acc[id] = 0.0f; occ[id] = 0; }
  if (id < 16) scal[id] = 0.0f;
}

// ---------------- x fp32 -> bf16 ----------------
__global__ __launch_bounds__(256) void k_cvt_x(const float* __restrict__ x, u16* __restrict__ xb) {
  int id = blockIdx.x * 256 + threadIdx.x;
  if (id < N_TOK * CDIM) xb[id] = f2bf(x[id]);
}

// ---------------- row L2-normalize (256 cols), one wave per row ----------------
__global__ __launch_bounds__(256) void k_norm_rows(const float* __restrict__ src,
                                                   float* __restrict__ dstF,
                                                   u16* __restrict__ dstH, int rows) {
  int wave = threadIdx.x >> 5, lane = threadIdx.x & 31;
  int n = blockIdx.x * 8 + wave;
  if (n >= rows) return;
  float v[8]; float ss = 0.0f;
  #pragma unroll
  for (int t = 0; t < 8; ++t) { v[t] = src[(size_t)n * MIDD + lane + 32 * t]; ss += v[t] * v[t]; }
  #pragma unroll
  for (int off = 16; off > 0; off >>= 1) ss += __shfl_xor(ss, off);
  float inv = 1.0f / fmaxf(sqrtf(ss), 1e-12f);
  #pragma unroll
  for (int t = 0; t < 8; ++t) {
    float o = v[t] * inv;
    dstF[(size_t)n * MIDD + lane + 32 * t] = o;
    dstH[(size_t)n * MIDD + lane + 32 * t] = f2bf(o);
  }
}

// ---------------- transpose W_down (768x256 f32) -> WT bf16 (256x768) ----------------
__global__ void k_transpose_w(const float* __restrict__ W, u16* __restrict__ wT) {
  __shared__ float sh[16][17];
  int k0 = blockIdx.x * 16, m0 = blockIdx.y * 16;
  sh[threadIdx.y][threadIdx.x] = W[(k0 + threadIdx.y) * MIDD + m0 + threadIdx.x];
  __syncthreads();
  wT[(m0 + threadIdx.y) * CDIM + k0 + threadIdx.x] = f2bf(sh[threadIdx.x][threadIdx.y]);
}

// ---------------- down projection: z_raw = x @ W + b (WMMA bf16) ----------------
__global__ __launch_bounds__(256) void k_down_proj(const u16* __restrict__ xb,
                                                   const u16* __restrict__ wT,
                                                   const float* __restrict__ bias,
                                                   float* __restrict__ z_raw) {
  int wave = threadIdx.x >> 5, lane = threadIdx.x & 31;
  int tile = blockIdx.x * 8 + wave;
  if (tile >= ROW_TILES) return;
  int r0 = tile * 16;
  int m = lane & 15, half = lane >> 4;
  int kbA = half * 16, kbB = half * 16, nn = lane & 15;
  for (int c = 0; c < MIDD / 16; ++c) {
    v8f acc = {0.f,0.f,0.f,0.f,0.f,0.f,0.f,0.f};
    #pragma unroll 4
    for (int s = 0; s < CDIM / 32; ++s) {
      const u16* ap = xb + (size_t)(r0 + m) * CDIM + s * 32 + kbA;
      v16bf A = cat16(*(const v8bf*)ap, *(const v8bf*)(ap + 8));
      const u16* bp = wT + (size_t)(c * 16 + nn) * CDIM + s * 32 + kbB;
      v16bf Bf = cat16(*(const v8bf*)bp, *(const v8bf*)(bp + 8));
      acc = wmma_bf16(A, Bf, acc);
    }
    float bcol = bias[c * 16 + nn];
    #pragma unroll
    for (int p = 0; p < 8; ++p) {
      int row = r0 + half * 8 + p;
      z_raw[(size_t)row * MIDD + c * 16 + nn] = acc[p] + bcol;
    }
  }
}

// ---------------- scorer pass A: online stats (max/argmax, 2-temp softmax sums) ----------------
__global__ __launch_bounds__(256) void k_score_a(const u16* __restrict__ znb,
                                                 const u16* __restrict__ enb,
                                                 float* __restrict__ ms_a, float* __restrict__ zt_a,
                                                 float* __restrict__ ze_a, float* __restrict__ su_a,
                                                 int* __restrict__ idx_a, int* __restrict__ occ) {
  int wave = threadIdx.x >> 5, lane = threadIdx.x & 31;
  int tile = blockIdx.x * 8 + wave;
  if (tile >= ROW_TILES) return;
  int r0 = tile * 16;
  int m = lane & 15, half = lane >> 4;
  int kbA = half * 16, kbB = half * 16, nn = lane & 15;

  v16bf Afr[8];
  #pragma unroll
  for (int s = 0; s < 8; ++s) {
    const u16* ap = znb + (size_t)(r0 + m) * MIDD + s * 32 + kbA;
    Afr[s] = cat16(*(const v8bf*)ap, *(const v8bf*)(ap + 8));
  }
  float mrun[8], Zt[8], Ze[8], Su[8]; int amax[8];
  #pragma unroll
  for (int p = 0; p < 8; ++p) { mrun[p] = -4.0f; Zt[p] = 0.f; Ze[p] = 0.f; Su[p] = 0.f; amax[p] = 0; }

  const u16* brow = enb + (size_t)nn * MIDD + kbB;
  for (int ct = 0; ct < COL_TILES; ++ct) {
    int c0 = ct * 16;
    const u16* bp = brow + (size_t)c0 * MIDD;
    if (ct + 1 < COL_TILES) __builtin_prefetch(bp + 16 * MIDD, 0, 1);
    v8f acc = {0.f,0.f,0.f,0.f,0.f,0.f,0.f,0.f};
    #pragma unroll
    for (int s = 0; s < 8; ++s)
      acc = wmma_bf16(Afr[s], cat16(*(const v8bf*)(bp + s * 32), *(const v8bf*)(bp + s * 32 + 8)), acc);
    int colg = c0 + nn;
    #pragma unroll
    for (int p = 0; p < 8; ++p) {
      float sv = acc[p];
      float mo = mrun[p];
      float mn = fmaxf(mo, sv);
      amax[p] = (sv > mo) ? colg : amax[p];
      float r  = __expf(TT2 * (mo - mn));
      float q  = __expf(TT2 * (sv - mn));
      float r7 = pow7(r), q7 = pow7(q);
      float t  = TE2 * (mo - mn);       // <= 0 (finite; m init is finite)
      float u  = TE2 * (sv - mn);
      float Zes = r7 * Ze[p];
      Zt[p] = Zt[p] * r + q;
      Su[p] = r7 * Su[p] + t * Zes + u * q7;
      Ze[p] = Zes + q7;
      mrun[p] = mn;
    }
  }
  // merge 16 lanes of each half (online-softmax state merge)
  #pragma unroll
  for (int p = 0; p < 8; ++p) {
    float m1 = mrun[p], Z1 = Zt[p], E1 = Ze[p], S1 = Su[p]; int i1 = amax[p];
    #pragma unroll
    for (int mask = 1; mask < 16; mask <<= 1) {
      float m2 = __shfl_xor(m1, mask), Z2 = __shfl_xor(Z1, mask);
      float E2 = __shfl_xor(E1, mask), S2 = __shfl_xor(S1, mask);
      int   i2 = __shfl_xor(i1, mask);
      float mm = fmaxf(m1, m2);
      float ra = __expf(TT2 * (m1 - mm)), rb = __expf(TT2 * (m2 - mm));
      float r7a = pow7(ra), r7b = pow7(rb);
      float ta = TE2 * (m1 - mm), tb = TE2 * (m2 - mm);
      Z1 = Z1 * ra + Z2 * rb;
      float Ea = r7a * E1, Eb = r7b * E2;
      S1 = r7a * S1 + ta * Ea + r7b * S2 + tb * Eb;
      E1 = Ea + Eb;
      i1 = (m2 > m1) ? i2 : i1;
      m1 = mm;
    }
    if ((lane & 15) == 0) {
      int row = r0 + half * 8 + p;
      ms_a[row] = m1; zt_a[row] = Z1; ze_a[row] = E1; su_a[row] = S1; idx_a[row] = i1;
      occ[i1] = 1;
    }
  }
}

// ---------------- gather z_q, vq loss, per-sample entropy term ----------------
__global__ __launch_bounds__(256) void k_gather(const float* __restrict__ enF,
                                                const float* __restrict__ znF,
                                                const int* __restrict__ idx_a,
                                                const float* __restrict__ ze_a,
                                                const float* __restrict__ su_a,
                                                float* __restrict__ out_zq, float* __restrict__ scal) {
  int wave = threadIdx.x >> 5, lane = threadIdx.x & 31;
  int n = blockIdx.x * 8 + wave;
  if (n >= N_TOK) return;
  int idx = idx_a[n];
  float ss = 0.0f;
  #pragma unroll
  for (int t = 0; t < 8; ++t) {
    int mc = lane + 32 * t;
    float zq = enF[(size_t)idx * MIDD + mc];
    float zv = znF[(size_t)n * MIDD + mc];
    out_zq[(size_t)n * MIDD + mc] = zq;
    float d = zq - zv; ss += d * d;
  }
  #pragma unroll
  for (int off = 16; off > 0; off >>= 1) ss += __shfl_xor(ss, off);
  if (lane == 0) {
    atomicAdd(&scal[0], ss);
    float Ze = ze_a[n], Su = su_a[n];
    atomicAdd(&scal[1], __logf(Ze) - Su / Ze);   // per-row entropy term
  }
}

// ---------------- scorer pass B: probs output + avg-prob accumulation ----------------
__global__ __launch_bounds__(256) void k_score_b(const u16* __restrict__ znb,
                                                 const u16* __restrict__ enb,
                                                 const float* __restrict__ ms_a,
                                                 const float* __restrict__ zt_a,
                                                 const float* __restrict__ ze_a,
                                                 float* __restrict__ probs, float* __restrict__ avg_acc) {
  int wave = threadIdx.x >> 5, lane = threadIdx.x & 31;
  int tile = blockIdx.x * 8 + wave;
  if (tile >= ROW_TILES) return;
  int r0 = tile * 16;
  int m = lane & 15, half = lane >> 4;
  int kbA = half * 16, kbB = half * 16, nn = lane & 15;

  v16bf Afr[8];
  #pragma unroll
  for (int s = 0; s < 8; ++s) {
    const u16* ap = znb + (size_t)(r0 + m) * MIDD + s * 32 + kbA;
    Afr[s] = cat16(*(const v8bf*)ap, *(const v8bf*)(ap + 8));
  }
  float msr[8], iZt[8], iZe[8];
  #pragma unroll
  for (int p = 0; p < 8; ++p) {
    int row = r0 + half * 8 + p;
    msr[p] = ms_a[row]; iZt[p] = 1.0f / zt_a[row]; iZe[p] = 1.0f / ze_a[row];
  }
  const u16* brow = enb + (size_t)nn * MIDD + kbB;
  for (int ct = 0; ct < COL_TILES; ++ct) {
    int c0 = ct * 16;
    const u16* bp = brow + (size_t)c0 * MIDD;
    if (ct + 1 < COL_TILES) __builtin_prefetch(bp + 16 * MIDD, 0, 1);
    v8f acc = {0.f,0.f,0.f,0.f,0.f,0.f,0.f,0.f};
    #pragma unroll
    for (int s = 0; s < 8; ++s)
      acc = wmma_bf16(Afr[s], cat16(*(const v8bf*)(bp + s * 32), *(const v8bf*)(bp + s * 32 + 8)), acc);
    float colsum = 0.0f;
    #pragma unroll
    for (int p = 0; p < 8; ++p) {
      float q = __expf(TT2 * (acc[p] - msr[p]));
      int row = r0 + half * 8 + p;
      probs[(size_t)row * KCB + c0 + nn] = q * iZt[p];
      colsum += pow7(q) * iZe[p];
    }
    colsum += __shfl_xor(colsum, 16);
    if (half == 0) atomicAdd(&avg_acc[c0 + nn], colsum);
  }
}

// ---------------- finalize scalars ----------------
__global__ __launch_bounds__(256) void k_finalize(const int* __restrict__ occ,
                                                  const float* __restrict__ avg_acc,
                                                  const float* __restrict__ scal,
                                                  float* __restrict__ outs) {
  __shared__ float sA[256], sB[256];
  int tid = threadIdx.x;
  float cnt = 0.0f, ae = 0.0f;
  for (int k = tid; k < KCB; k += 256) {
    cnt += (occ[k] != 0) ? 1.0f : 0.0f;
    float a = avg_acc[k] * (1.0f / (float)N_TOK);
    ae += a * __logf(a + 1e-5f);
  }
  sA[tid] = cnt; sB[tid] = ae; __syncthreads();
  for (int off = 128; off > 0; off >>= 1) {
    if (tid < off) { sA[tid] += sA[tid + off]; sB[tid] += sB[tid + off]; }
    __syncthreads();
  }
  if (tid == 0) {
    float usage = sA[0] / (float)KCB;
    float vq = scal[0] / (float)(N_TOK * MIDD);
    float sample_entropy = scal[1] / (float)N_TOK;
    float entropy_loss = sample_entropy + sB[0];   // se - (-sum a*log(a+eps))
    outs[0] = usage; outs[1] = vq; outs[2] = vq; outs[3] = entropy_loss;
  }
}

extern "C" void kernel_launch(void* const* d_in, const int* in_sizes, int n_in,
                              void* d_out, int out_size, void* d_ws, size_t ws_size,
                              hipStream_t stream) {
  const float* x   = (const float*)d_in[0];   // [7200,768]
  const float* W   = (const float*)d_in[1];   // [768,256]
  const float* bia = (const float*)d_in[2];   // [256]
  const float* emb = (const float*)d_in[3];   // [8912,256]
  float* out = (float*)d_out;
  float* out_zq    = out;                       // 7200*256
  float* out_probs = out + (size_t)N_TOK * MIDD;               // 7200*8912
  float* out_scal  = out + (size_t)N_TOK * MIDD + (size_t)N_TOK * KCB; // 4 scalars

  char* p = (char*)d_ws;
  auto alloc = [&](size_t bytes) { void* r = (void*)p; p += (bytes + 255) & ~(size_t)255; return r; };
  u16*   x_bf   = (u16*)  alloc((size_t)N_TOK * CDIM * 2);
  float* zn_f   = (float*)alloc((size_t)N_TOK * MIDD * 4);
  u16*   zn_bf  = (u16*)  alloc((size_t)N_TOK * MIDD * 2);
  float* en_f   = (float*)alloc((size_t)KCB * MIDD * 4);
  u16*   en_bf  = (u16*)  alloc((size_t)KCB * MIDD * 2);
  u16*   wT     = (u16*)  alloc((size_t)CDIM * MIDD * 2);
  float* ms_a   = (float*)alloc((size_t)N_TOK * 4);
  float* zt_a   = (float*)alloc((size_t)N_TOK * 4);
  float* ze_a   = (float*)alloc((size_t)N_TOK * 4);
  float* su_a   = (float*)alloc((size_t)N_TOK * 4);
  int*   idx_a  = (int*)  alloc((size_t)N_TOK * 4);
  float* avg_ac = (float*)alloc((size_t)KCB * 4);
  int*   occ    = (int*)  alloc((size_t)KCB * 4);
  float* scal   = (float*)alloc(64);

  k_init<<<(KCB + 255) / 256, 256, 0, stream>>>(avg_ac, occ, scal);
  k_cvt_x<<<(N_TOK * CDIM + 255) / 256, 256, 0, stream>>>(x, x_bf);
  k_norm_rows<<<(KCB + 7) / 8, 256, 0, stream>>>(emb, en_f, en_bf, KCB);
  k_transpose_w<<<dim3(CDIM / 16, MIDD / 16), dim3(16, 16), 0, stream>>>(W, wT);
  k_down_proj<<<(ROW_TILES + 7) / 8, 256, 0, stream>>>(x_bf, wT, bia, zn_f);
  k_norm_rows<<<(N_TOK + 7) / 8, 256, 0, stream>>>(zn_f, zn_f, zn_bf, N_TOK);
  k_score_a<<<(ROW_TILES + 7) / 8, 256, 0, stream>>>(zn_bf, en_bf, ms_a, zt_a, ze_a, su_a, idx_a, occ);
  k_gather<<<(N_TOK + 7) / 8, 256, 0, stream>>>(en_f, zn_f, idx_a, ze_a, su_a, out_zq, scal);
  k_score_b<<<(ROW_TILES + 7) / 8, 256, 0, stream>>>(zn_bf, en_bf, ms_a, zt_a, ze_a, out_probs, avg_ac);
  k_finalize<<<1, 256, 0, stream>>>(occ, avg_ac, scal, out_scal);
}
